// GraphConvolution_7464653161211
// MI455X (gfx1250) — matrix operand order repaired and checked by reading, not compiled, not run on hardware
//
#include <hip/hip_runtime.h>
#include <hip/hip_bf16.h>

// ---------------------------------------------------------------------------
// GraphConvolution attention for MI455X (gfx1250, wave32, WMMA).
//   out = softmax(topk(softmax(k q^T))) @ (x @ (Wv^T @ weight)) + bias
// B=8 S=2048 C=256 D=64 OUT=256 KTOP=1365. f16 WMMA with f32 accumulate.
// ---------------------------------------------------------------------------

typedef __attribute__((ext_vector_type(16))) _Float16 v16h;
typedef __attribute__((ext_vector_type(8)))  _Float16 v8h;
typedef __attribute__((ext_vector_type(8)))  float    v8f;

#define NB    8
#define NS    2048
#define NC    256
#define ND    64
#define NOUT  256
#define KTOP  1365
#define NROWS (NB * NS)          // 16384 flattened (b,s) rows
#define NALL  384                // 64 q + 64 k + 256 support columns

__device__ __forceinline__ v8f wmma_f16(v16h a, v16h b, v8f c) {
  // D = A(16x32 f16) x B(32x16 f16) + C(16x16 f32)
  return __builtin_amdgcn_wmma_f32_16x16x32_f16(
      /*neg_a=*/false, a, /*neg_b=*/false, b,
      /*c_mod=*/(short)0, c, /*reuse_a=*/false, /*reuse_b=*/false);
}

// B fragment: lane (n=lane&15, hi=lane>>4) holds B[K = hi*16 .. hi*16+15][n].
// With B's column n stored K-contiguous (row-major weight / transposed support)
// this is one 32-byte contiguous load.
__device__ __forceinline__ v16h load_bfrag(const _Float16* p) {
  v8h c0 = *reinterpret_cast<const v8h*>(p);
  v8h c1 = *reinterpret_cast<const v8h*>(p + 8);
  v16h r;
#pragma unroll
  for (int i = 0; i < 8; ++i) { r[i] = c0[i]; r[i + 8] = c1[i]; }
  return r;
}

// A fragment from f16 row-major: lane (m=lane&15, hi) holds
// A[m][hi*16 .. hi*16+15].  p = &A[m][hi*16] of the
// current 32-wide K slab; second chunk at p+8.
__device__ __forceinline__ v16h load_afrag_f16(const _Float16* p) {
  v8h c0 = *reinterpret_cast<const v8h*>(p);
  v8h c1 = *reinterpret_cast<const v8h*>(p + 8);
  v16h r;
#pragma unroll
  for (int i = 0; i < 8; ++i) { r[i] = c0[i]; r[i + 8] = c1[i]; }
  return r;
}

// Same A layout but converting f32 (global x, or LDS adj rows) -> f16.
__device__ __forceinline__ v16h load_afrag_f32(const float* p) {
  v16h r;
#pragma unroll
  for (int i = 0; i < 8; ++i) {
    r[i]     = (_Float16)p[i];
    r[i + 8] = (_Float16)p[8 + i];
  }
  return r;
}

__device__ __forceinline__ float wave_max(float v) {
#pragma unroll
  for (int o = 16; o > 0; o >>= 1) v = fmaxf(v, __shfl_xor(v, o, 32));
  return v;
}
__device__ __forceinline__ float wave_sum(float v) {
#pragma unroll
  for (int o = 16; o > 0; o >>= 1) v += __shfl_xor(v, o, 32);
  return v;
}
__device__ __forceinline__ int wave_isum(int v) {
#pragma unroll
  for (int o = 16; o > 0; o >>= 1) v += __shfl_xor(v, o, 32);
  return v;
}

// ---------------------------------------------------------------------------
// Kernel 0: fold weights.  Wall[row][c] (f16, 384x256):
//   row   0..63  = Wq[row]          row 64..127 = Wk[row-64]
//   row 128..383 = W2^T[u] with W2[c][u] = sum_o Wv[o][c] * weight[o][u]
// ball[row] = matching bias (bq | bk | bv @ weight).
// ---------------------------------------------------------------------------
__global__ __launch_bounds__(256) void prep_kernel(
    const float* __restrict__ Wq, const float* __restrict__ bq,
    const float* __restrict__ Wk, const float* __restrict__ bk,
    const float* __restrict__ Wv, const float* __restrict__ bv,
    const float* __restrict__ weight,
    _Float16* __restrict__ Wall, float* __restrict__ ball) {
  int row = blockIdx.x;      // 0..383
  int c   = threadIdx.x;     // 0..255
  float v;
  if (row < 64) {
    v = Wq[row * NC + c];
  } else if (row < 128) {
    v = Wk[(row - 64) * NC + c];
  } else {
    int u = row - 128;
    float s = 0.f;
#pragma unroll 4
    for (int o = 0; o < NOUT; ++o) s += Wv[o * NC + c] * weight[o * NOUT + u];
    v = s;
  }
  Wall[row * NC + c] = (_Float16)v;
  if (c == 0) {
    float bs;
    if (row < 64)        bs = bq[row];
    else if (row < 128)  bs = bk[row - 64];
    else {
      int u = row - 128; float s = 0.f;
      for (int o = 0; o < NOUT; ++o) s += bv[o] * weight[o * NOUT + u];
      bs = s;
    }
    ball[row] = bs;
  }
}

// ---------------------------------------------------------------------------
// Kernel 1: fused projections.  One wave per (16-row, 16-col) tile of the
// [16384 x 384] output = x[16384x256] @ Wall^T, K=256 in 8 WMMA steps.
// Columns 0..63 -> qh (row-major f16), 64..127 -> kh, 128..383 -> support,
// stored TRANSPOSED as supT[b][u][s] so kernel 2's B fragments are contiguous.
// ---------------------------------------------------------------------------
__global__ __launch_bounds__(32) void proj_kernel(
    const float* __restrict__ x, const _Float16* __restrict__ Wall,
    const float* __restrict__ ball,
    _Float16* __restrict__ qh, _Float16* __restrict__ kh,
    _Float16* __restrict__ supT) {
  int rt = blockIdx.x;            // 0..1023 row tile
  int ct = blockIdx.y;            // 0..23   col tile
  int lane = threadIdx.x;
  int n  = lane & 15;
  int hi = lane >> 4;
  int row0 = rt * 16, n0 = ct * 16;

  v8f acc = {};
#pragma unroll
  for (int kb = 0; kb < 8; ++kb) {
    int kk = kb * 32;
    v16h a = load_afrag_f32(x + (size_t)(row0 + n) * NC + kk + hi * 16);
    v16h b = load_bfrag(Wall + (size_t)(n0 + n) * NC + kk + hi * 16);
    acc = wmma_f16(a, b, acc);
  }
  float bb = ball[n0 + n];
#pragma unroll
  for (int r = 0; r < 8; ++r) {
    int g = row0 + r + 8 * hi;              // global (b,s) row
    float val = acc[r] + bb;
    if (n0 < 64) {
      qh[(size_t)g * ND + n0 + n] = (_Float16)val;
    } else if (n0 < 128) {
      kh[(size_t)g * ND + (n0 - 64) + n] = (_Float16)val;
    } else {
      int b = g >> 11, s = g & (NS - 1);
      int u = (n0 - 128) + n;
      supT[((size_t)(b * NOUT + u)) * NS + s] = (_Float16)val;
    }
  }
}

// ---------------------------------------------------------------------------
// Kernel 2: per 16-row tile (block = 8 waves):
//  P1: scores[16][2048] = k_tile @ q^T (WMMA, K=64) -> 128 KB dynamic LDS
//  P2-4: per row (one wave owns 2 rows): softmax; top-K threshold by at most 30-step
//        bisection on the f32 bit pattern (positive floats are bit-monotone);
//        masked re-softmax written back in place.  Ties at the threshold are
//        all kept (reference top_k breaks ties by index).
//  P5: out_tile = adj2[16][2048] @ support[2048][256] + bias (WMMA, 64 K-steps,
//      A converted f32->f16 straight from LDS, B contiguous from supT).
// ---------------------------------------------------------------------------
__global__ __launch_bounds__(256) void attn_kernel(
    const _Float16* __restrict__ qh, const _Float16* __restrict__ kh,
    const _Float16* __restrict__ supT, const float* __restrict__ bias,
    float* __restrict__ out) {
  extern __shared__ float sm[];                 // [16][2048] scores/adj2
  int b  = blockIdx.x >> 7;                     // batch
  int it = blockIdx.x & 127;                    // i tile
  int i0 = it * 16;
  int tid = threadIdx.x, w = tid >> 5, lane = tid & 31;
  int n = lane & 15, hi = lane >> 4;

  // ---- Phase 1: scores = k_tile . q^T --------------------------------------
  const _Float16* kbase = kh + (size_t)(b * NS + i0) * ND;
  v16h ak0 = load_afrag_f16(kbase + (size_t)n * ND +  0 + hi * 16);
  v16h ak1 = load_afrag_f16(kbase + (size_t)n * ND + 32 + hi * 16);
  for (int jt = w; jt < 128; jt += 8) {
    int j0 = jt * 16;
    const _Float16* qrow = qh + (size_t)(b * NS + j0 + n) * ND;
    v8f acc = {};
    acc = wmma_f16(ak0, load_bfrag(qrow +  0 + hi * 16), acc);
    acc = wmma_f16(ak1, load_bfrag(qrow + 32 + hi * 16), acc);
#pragma unroll
    for (int r = 0; r < 8; ++r)
      sm[(size_t)(r + 8 * hi) * NS + j0 + n] = acc[r];
  }
  __syncthreads();

  // ---- Phases 2-4: softmax, top-k threshold, masked re-softmax -------------
  for (int rr = 0; rr < 2; ++rr) {
    float* row = sm + (size_t)(w * 2 + rr) * NS;
    float mx = -3.4e38f;
    for (int j = lane; j < NS; j += 32) mx = fmaxf(mx, row[j]);
    mx = wave_max(mx);
    float s = 0.f;
    for (int j = lane; j < NS; j += 32) {
      float e = __expf(row[j] - mx);
      row[j] = e; s += e;
    }
    s = wave_sum(s);
    float inv = 1.f / s, pmax = 0.f;
    for (int j = lane; j < NS; j += 32) {
      float p = row[j] * inv;
      row[j] = p; pmax = fmaxf(pmax, p);
    }
    pmax = wave_max(pmax);
    // bisection for the KTOP-th largest probability (bits of positive f32
    // compare like the values themselves).
    unsigned lo = 0u, hi_ = 0x3F800001u;            // p <= 1.0
    while (lo < hi_) {
      unsigned mid = lo + ((hi_ - lo + 1u) >> 1);
      int cnt = 0;
      for (int j = lane; j < NS; j += 32)
        cnt += (__float_as_uint(row[j]) >= mid) ? 1 : 0;
      cnt = wave_isum(cnt);
      if (cnt >= KTOP) lo = mid; else hi_ = mid - 1u;
    }
    float thr = __uint_as_float(lo);
    float s2 = 0.f;
    for (int j = lane; j < NS; j += 32) {
      float p = row[j];
      s2 += (p >= thr) ? __expf(p - pmax) : 0.f;
    }
    s2 = wave_sum(s2);
    float inv2 = 1.f / s2;
    for (int j = lane; j < NS; j += 32) {
      float p = row[j];
      row[j] = (p >= thr) ? __expf(p - pmax) * inv2 : 0.f;
    }
  }
  __syncthreads();

  // ---- Phase 5: out_tile = adj2 @ support + bias ---------------------------
  const _Float16* sbase = supT + (size_t)b * NOUT * NS;
  const _Float16* brow0 = sbase + (size_t)(w * 16 + n) * NS;
  const _Float16* brow1 = sbase + (size_t)(128 + w * 16 + n) * NS;
  v8f o0 = {}, o1 = {};
  for (int ks = 0; ks < 64; ++ks) {
    int kk = ks * 32;
    __builtin_prefetch(brow0 + kk + 128, 0, 1);   // global_prefetch_b8
    __builtin_prefetch(brow1 + kk + 128, 0, 1);
    v16h a  = load_afrag_f32(sm + (size_t)n * NS + kk + hi * 16);
    v16h b0 = load_bfrag(brow0 + kk + hi * 16);
    v16h b1 = load_bfrag(brow1 + kk + hi * 16);
    o0 = wmma_f16(a, b0, o0);
    o1 = wmma_f16(a, b1, o1);
  }
  float bi0 = bias[w * 16 + n];
  float bi1 = bias[128 + w * 16 + n];
#pragma unroll
  for (int r = 0; r < 8; ++r) {
    size_t orow = ((size_t)(b * NS + i0 + r + 8 * hi)) * NOUT;
    out[orow + w * 16 + n]       = o0[r] + bi0;
    out[orow + 128 + w * 16 + n] = o1[r] + bi1;
  }
}

// ---------------------------------------------------------------------------
// Host launcher.  Workspace layout (needs ~12.3 MB):
//   [0)        Wall  f16 384*256            = 196608 B
//   [196608)   ball  f32 384                =   1536 B
//   [262144)   qh    f16 16384*64           =   2 MB
//   [2359296)  kh    f16 16384*64           =   2 MB
//   [4456448)  supT  f16 8*256*2048         =   8 MB
// ---------------------------------------------------------------------------
extern "C" void kernel_launch(void* const* d_in, const int* in_sizes, int n_in,
                              void* d_out, int out_size, void* d_ws, size_t ws_size,
                              hipStream_t stream) {
  const float* x      = (const float*)d_in[0];
  const float* Wq     = (const float*)d_in[1];
  const float* bq     = (const float*)d_in[2];
  const float* Wk     = (const float*)d_in[3];
  const float* bk     = (const float*)d_in[4];
  const float* Wv     = (const float*)d_in[5];
  const float* bv     = (const float*)d_in[6];
  const float* weight = (const float*)d_in[7];
  const float* bias   = (const float*)d_in[8];
  float* out = (float*)d_out;

  char* ws = (char*)d_ws;
  _Float16* Wall = (_Float16*)(ws + 0);
  float*    ball = (float*)   (ws + 196608);
  _Float16* qh   = (_Float16*)(ws + 262144);
  _Float16* kh   = (_Float16*)(ws + 262144 + 2097152);
  _Float16* supT = (_Float16*)(ws + 262144 + 2 * 2097152);

  prep_kernel<<<dim3(NALL), dim3(256), 0, stream>>>(Wq, bq, Wk, bk, Wv, bv,
                                                    weight, Wall, ball);
  proj_kernel<<<dim3(NROWS / 16, NALL / 16), dim3(32), 0, stream>>>(
      x, Wall, ball, qh, kh, supT);
  attn_kernel<<<dim3(NB * (NS / 16)), dim3(256), 16 * NS * sizeof(float),
                stream>>>(qh, kh, supT, bias, out);
}
